// ExpendMemoryUnit_40389872452150
// MI455X (gfx1250) — matrix operand order repaired and checked by reading, not compiled, not run on hardware
//
#include <hip/hip_runtime.h>

// ---------------------------------------------------------------------------
// Types for CDNA5 WMMA (wave32, v_wmma_f32_16x16x32_bf16)
// ---------------------------------------------------------------------------
typedef __attribute__((ext_vector_type(16))) __bf16          bf16x16;
typedef __attribute__((ext_vector_type(8)))  float           f32x8;
typedef __attribute__((ext_vector_type(8)))  unsigned short  u16x8;

union Frag { bf16x16 v; u16x8 h[2]; };
union U8p  { u16x8 v; unsigned short s[8]; };

__device__ __forceinline__ unsigned short f2bf(float f) {
  union { float f; unsigned u; } c; c.f = f;
  return (unsigned short)((c.u + 0x7FFFu + ((c.u >> 16) & 1u)) >> 16);
}

#define WMMA_BF16(A, B, Cacc) \
  __builtin_amdgcn_wmma_f32_16x16x32_bf16(false, (A), false, (B), (short)0, (Cacc), false, false)

// CDNA5 async global->LDS copy (ASYNCcnt-tracked; bypasses VGPRs).
// LDS byte address = low 32 bits of the flat shared pointer (LDS aperture).
__device__ __forceinline__ void async_g2l(unsigned lds, const void* g) {
  asm volatile("global_load_async_to_lds_b128 %0, %1, off"
               :: "v"(lds), "v"(g) : "memory");
}
__device__ __forceinline__ void wait_async0() {
  asm volatile("s_wait_asynccnt 0x0" ::: "memory");
}
__device__ __forceinline__ unsigned lds_addr(const void* p) {
  return (unsigned)(unsigned long long)p;
}

// ---------------------------------------------------------------------------
// Generic tiled WMMA GEMM:  D[b][M][N] = A[b?][M][K] * Bt[b?][N][K]^T
// A, Bt are bf16 (as ushort). Epilogue: +bias, +noise, activation, +addsrc,
// optional f32 out [b][M][N] and/or transposed bf16 out [b][N][M].
// M,N multiples of 64; K multiple of 32. Block = 128 threads (4 waves, 2x2).
// Double-buffered LDS tiles filled by async global->LDS DMA.
// ---------------------------------------------------------------------------
#define GSTR 40   // LDS row stride in ushorts (80B, 16B-aligned chunks)
#define TILEB (64 * GSTR * 2)  // one tile buffer in bytes

__global__ __launch_bounds__(128)
void k_gemm(const unsigned short* __restrict__ A,  long aB,
            const unsigned short* __restrict__ Bt, long bB,
            const float* __restrict__ bias, int biasB,
            int act,
            const float* __restrict__ noise, const float* __restrict__ noiseW,
            const float* __restrict__ addsrc,
            float* __restrict__ outF32,
            unsigned short* __restrict__ outBf16T,
            int M, int N, int K)
{
  __shared__ unsigned short sA[2][64 * GSTR];
  __shared__ unsigned short sB[2][64 * GSTR];

  const int b  = blockIdx.z;
  const int m0 = blockIdx.y * 64, n0 = blockIdx.x * 64;
  const unsigned short* Ab = A  + (long)b * aB;
  const unsigned short* Bb = Bt + (long)b * bB;

  const int tid = threadIdx.x;
  const int wave = tid >> 5, lane = tid & 31;
  const int wr = wave >> 1, wc = wave & 1;
  const int lm = lane & 15, lh = lane >> 4;

  // Per-thread staging slot: each of 128 threads owns one 32B chunk per tile.
  const int r = tid >> 1, ho = (tid & 1) * 16;
  const unsigned short* gA = Ab + (long)(m0 + r) * K + ho;
  const unsigned short* gB = Bb + (long)(n0 + r) * K + ho;
  const unsigned ldsA0 = lds_addr(&sA[0][r * GSTR + ho]);
  const unsigned ldsB0 = lds_addr(&sB[0][r * GSTR + ho]);

  auto issue = [&](int kk, int bufi) {
    const unsigned la = ldsA0 + bufi * TILEB;
    const unsigned lb = ldsB0 + bufi * TILEB;
    async_g2l(la,      gA + kk);
    async_g2l(la + 16, gA + kk + 8);
    async_g2l(lb,      gB + kk);
    async_g2l(lb + 16, gB + kk + 8);
  };

  f32x8 acc[2][2];
  f32x8 zero = {};
  for (int i = 0; i < 2; i++) for (int j = 0; j < 2; j++) acc[i][j] = zero;

  issue(0, 0);
  wait_async0();
  __syncthreads();

  int buf = 0;
  for (int kk = 0; kk < K; kk += 32) {
    if (kk + 32 < K) issue(kk + 32, buf ^ 1);   // prefetch next tile (async DMA)

    const unsigned short* cA = sA[buf];
    const unsigned short* cB = sB[buf];
    Frag af[2], bf_[2];
#pragma unroll
    for (int t = 0; t < 2; t++) {
      const int row = wr * 32 + t * 16 + lm;   // A: K chunks at 8*lh and 16+8*lh
      af[t].h[0] = *(const u16x8*)&cA[row * GSTR + 8 * lh];
      af[t].h[1] = *(const u16x8*)&cA[row * GSTR + 16 + 8 * lh];
      const int col = wc * 32 + t * 16 + lm;   // B: K chunks at 8*lh and 16+8*lh
      bf_[t].h[0] = *(const u16x8*)&cB[col * GSTR + 8 * lh];
      bf_[t].h[1] = *(const u16x8*)&cB[col * GSTR + 16 + 8 * lh];
    }
#pragma unroll
    for (int i = 0; i < 2; i++)
#pragma unroll
      for (int j = 0; j < 2; j++)
        acc[i][j] = WMMA_BF16(af[i].v, bf_[j].v, acc[i][j]);

    wait_async0();      // my prefetch arrived in LDS
    __syncthreads();    // everyone's prefetch arrived; everyone done reading cur
    buf ^= 1;
  }

  // Epilogue
  const float nw = noise ? noiseW[0] : 0.f;
  const float* biasb = bias + (long)(biasB ? b * biasB : 0);
#pragma unroll
  for (int i = 0; i < 2; i++)
#pragma unroll
    for (int j = 0; j < 2; j++) {
      const int mbase = m0 + wr * 32 + i * 16;
      const int n = n0 + wc * 32 + j * 16 + lm;
      U8p packed;
#pragma unroll
      for (int rr = 0; rr < 8; rr++) {
        const int m = mbase + rr + 8 * lh;
        float x = acc[i][j][rr] + biasb[m];
        if (noise) x += nw * noise[(long)b * N + n];
        if (act == 1)      x = fmaxf(x, 0.f);
        else if (act == 2) x = (x > 0.f ? x : 0.2f * x) * 1.41421356237f;
        if (addsrc) x += addsrc[((long)b * M + m) * (long)N + n];
        if (outF32) outF32[((long)b * M + m) * (long)N + n] = x;
        packed.s[rr] = f2bf(x);
      }
      if (outBf16T)
        *(u16x8*)&outBf16T[((long)b * N + n) * (long)M + mbase + 8 * lh] = packed.v;
    }
}

// ---------------------------------------------------------------------------
// Fused flash attention:  residual = softmax(Q K^T / 4) V ;
// writes diffin[b][p][c] = bf16(feat[b][c][p] - residual[p][c]).
// Block: 128 threads (4 waves), 64 query rows; loops over 4096 keys in 64-blocks.
// K/V tiles filled via async global->LDS DMA.
// ---------------------------------------------------------------------------
__global__ __launch_bounds__(128)
void k_attn(const unsigned short* __restrict__ qtok,  // [b][4096][128]
            const unsigned short* __restrict__ ktok,  // [b][4096][128]
            const unsigned short* __restrict__ vmat,  // [b][128][4096]
            const float* __restrict__ feat,           // [b][128][4096]
            unsigned short* __restrict__ diffin)      // [b][4096][128]
{
  __shared__ unsigned short sQ[64 * 136];
  __shared__ unsigned short sK[64 * 136];
  __shared__ unsigned short sV[128 * 72];
  __shared__ unsigned short sP[64 * 72];

  const int b = blockIdx.y;
  const int q0 = blockIdx.x * 64;
  const int tid = threadIdx.x, wave = tid >> 5, lane = tid & 31;
  const int lm = lane & 15, lh = lane >> 4;

  {
    const unsigned short* src = qtok + ((long)b * 4096 + q0) * 128;
    for (int idx = tid; idx < 64 * 16; idx += 128) {
      const int rr = idx >> 4, ch = idx & 15;
      async_g2l(lds_addr(&sQ[rr * 136 + ch * 8]), &src[rr * 128 + ch * 8]);
    }
    wait_async0();
  }
  __syncthreads();

  Frag qf[4];
  {
    const int row = wave * 16 + lm;
#pragma unroll
    for (int ks = 0; ks < 4; ks++) {
      qf[ks].h[0] = *(const u16x8*)&sQ[row * 136 + ks * 32 + 8 * lh];
      qf[ks].h[1] = *(const u16x8*)&sQ[row * 136 + ks * 32 + 16 + 8 * lh];
    }
  }

  f32x8 zero = {};
  f32x8 oacc[8];
#pragma unroll
  for (int i = 0; i < 8; i++) oacc[i] = zero;
  float mrun[8], lrun[8];
#pragma unroll
  for (int rr = 0; rr < 8; rr++) { mrun[rr] = -1e30f; lrun[rr] = 0.f; }

  for (int j0 = 0; j0 < 4096; j0 += 64) {
    __syncthreads();
    {
      const unsigned short* src = ktok + ((long)b * 4096 + j0) * 128;
      for (int idx = tid; idx < 64 * 16; idx += 128) {
        const int rr = idx >> 4, ch = idx & 15;
        async_g2l(lds_addr(&sK[rr * 136 + ch * 8]), &src[rr * 128 + ch * 8]);
      }
      const unsigned short* vs = vmat + (long)b * 128 * 4096 + j0;
      for (int idx = tid; idx < 128 * 8; idx += 128) {
        const int d = idx >> 3, ch = idx & 7;
        async_g2l(lds_addr(&sV[d * 72 + ch * 8]), &vs[(long)d * 4096 + ch * 8]);
      }
      wait_async0();
    }
    __syncthreads();

    // S = Q K^T (16x64 per wave)
    f32x8 sacc[4];
#pragma unroll
    for (int t = 0; t < 4; t++) sacc[t] = zero;
#pragma unroll
    for (int ks = 0; ks < 4; ks++)
#pragma unroll
      for (int t = 0; t < 4; t++) {
        Frag kf;
        const int col = t * 16 + lm;
        kf.h[0] = *(const u16x8*)&sK[col * 136 + ks * 32 + 8 * lh];
        kf.h[1] = *(const u16x8*)&sK[col * 136 + ks * 32 + 16 + 8 * lh];
        sacc[t] = WMMA_BF16(qf[ks].v, kf.v, sacc[t]);
      }

    // Online softmax (row r of half lh; reduce across 16 lanes of the half)
#pragma unroll
    for (int rr = 0; rr < 8; rr++) {
      float mx = -1e30f;
#pragma unroll
      for (int t = 0; t < 4; t++) mx = fmaxf(mx, sacc[t][rr] * 0.25f);
      for (int m = 1; m < 16; m <<= 1) mx = fmaxf(mx, __shfl_xor(mx, m, 32));
      const float mnew = fmaxf(mrun[rr], mx);
      const float corr = __expf(mrun[rr] - mnew);
      mrun[rr] = mnew;
      float sum = 0.f;
#pragma unroll
      for (int t = 0; t < 4; t++) {
        const float p = __expf(sacc[t][rr] * 0.25f - mnew);
        sacc[t][rr] = p; sum += p;
      }
      for (int m = 1; m < 16; m <<= 1) sum += __shfl_xor(sum, m, 32);
      lrun[rr] = lrun[rr] * corr + sum;
#pragma unroll
      for (int nt = 0; nt < 8; nt++) oacc[nt][rr] *= corr;
    }

    // P -> LDS (bf16), then O += P V
#pragma unroll
    for (int t = 0; t < 4; t++)
#pragma unroll
      for (int rr = 0; rr < 8; rr++)
        sP[(wave * 16 + rr + 8 * lh) * 72 + t * 16 + lm] = f2bf(sacc[t][rr]);
    __syncthreads();

#pragma unroll
    for (int ks = 0; ks < 2; ks++) {
      Frag pf;
      const int prow = wave * 16 + lm;
      pf.h[0] = *(const u16x8*)&sP[prow * 72 + ks * 32 + 8 * lh];
      pf.h[1] = *(const u16x8*)&sP[prow * 72 + ks * 32 + 16 + 8 * lh];
#pragma unroll
      for (int nt = 0; nt < 8; nt++) {
        Frag vf;
        const int d = nt * 16 + lm;
        vf.h[0] = *(const u16x8*)&sV[d * 72 + ks * 32 + 8 * lh];
        vf.h[1] = *(const u16x8*)&sV[d * 72 + ks * 32 + 16 + 8 * lh];
        oacc[nt] = WMMA_BF16(pf.v, vf.v, oacc[nt]);
      }
    }
  }

  // diffin = feat - O/l   (token layout [p][c])
#pragma unroll
  for (int rr = 0; rr < 8; rr++) {
    const float invl = 1.f / lrun[rr];
    const int prow = q0 + wave * 16 + rr + 8 * lh;
#pragma unroll
    for (int nt = 0; nt < 8; nt++) {
      const int d = nt * 16 + lm;
      const float x = feat[((long)b * 128 + d) * 4096 + prow] - oacc[nt][rr] * invl;
      diffin[((long)b * 4096 + prow) * 128 + d] = f2bf(x);
    }
  }
}

// ---------------------------------------------------------------------------
// Small / glue kernels
// ---------------------------------------------------------------------------
__global__ void k_copy_even(const float* __restrict__ f, float* __restrict__ out) {
  const long i = (long)blockIdx.x * 256 + threadIdx.x;
  if (i >= 4L * 128 * 4096) return;
  const long p = i & 4095, c = (i >> 12) & 127, b = i >> 19;
  out[((b * 256 + 2 * c) * 4096) + p] = f[((b * 256 + 2 * c) * 4096) + p];
}

__global__ void k_f2bf(const float* __restrict__ s, unsigned short* __restrict__ d, long n) {
  const long i = (long)blockIdx.x * 256 + threadIdx.x;
  if (i < n) d[i] = f2bf(s[i]);
}

__global__ void k_pack_cs(const float* __restrict__ f, unsigned short* __restrict__ t) {
  const int b = blockIdx.y, n = blockIdx.x, ci = threadIdx.x;
  t[((long)b * 4096 + n) * 128 + ci] = f2bf(f[(((long)b * 256 + 2 * ci + 1) * 4096) + n]);
}

__global__ void k_im2col(const float* __restrict__ src, long bStride,
                         unsigned short* __restrict__ dst, long dB, int C) {
  const int b = blockIdx.z, n = blockIdx.x;
  const int h = n >> 6, w = n & 63;
  const float* s = src + (long)b * bStride;
  unsigned short* d = dst + (long)b * dB + (long)n * (C * 9);
  for (int ci = threadIdx.x; ci < C; ci += blockDim.x) {
    const float* sc = s + (long)ci * 4096;
    unsigned short* dd = d + ci * 9;
#pragma unroll
    for (int kh = 0; kh < 3; kh++)
#pragma unroll
      for (int kw = 0; kw < 3; kw++) {
        const int hh = h + kh - 1, ww = w + kw - 1;
        const float v = (hh >= 0 && hh < 64 && ww >= 0 && ww < 64) ? sc[hh * 64 + ww] : 0.f;
        dd[kh * 3 + kw] = f2bf(v);
      }
  }
}

__global__ void k_rowmean(const float* __restrict__ src, float* __restrict__ dst) {
  __shared__ float red[256];
  const long row = blockIdx.x;
  const float* s = src + row * 4096;
  float acc = 0.f;
  for (int i = threadIdx.x; i < 4096; i += 256) acc += s[i];
  red[threadIdx.x] = acc; __syncthreads();
  for (int o = 128; o > 0; o >>= 1) {
    if ((int)threadIdx.x < o) red[threadIdx.x] += red[threadIdx.x + o];
    __syncthreads();
  }
  if (threadIdx.x == 0) dst[row] = red[0] * (1.f / 4096.f);
}

__global__ void k_build_sc(const float* __restrict__ scode, const float* __restrict__ kps,
                           float* __restrict__ sc) {
  const int b = blockIdx.x, t = threadIdx.x;
  if (t < 128) sc[b * 158 + t] = scode[b * 128 + t];
  else if (t < 158) sc[b * 158 + t] = kps[b * 30 + (t - 128)];
}

__global__ void k_mlp(const float* __restrict__ sc,
                      const float* __restrict__ w1, const float* __restrict__ b1,
                      const float* __restrict__ w2, const float* __restrict__ b2,
                      const float* __restrict__ modw, const float* __restrict__ modb,
                      float* __restrict__ style) {
  __shared__ float s_sc[158], s_h1[128], s_code[512];
  const int b = blockIdx.x, t = threadIdx.x;
  if (t < 158) s_sc[t] = sc[b * 158 + t];
  __syncthreads();
  if (t < 128) {
    float a = b1[t];
    for (int k = 0; k < 158; k++) a += w1[t * 158 + k] * s_sc[k];
    s_h1[t] = fmaxf(a, 0.f);
  }
  __syncthreads();
  {
    float a = b2[t];
    for (int k = 0; k < 128; k++) a += w2[t * 128 + k] * s_h1[k];
    s_code[t] = a;
  }
  __syncthreads();
  if (t < 128) {
    const float inv = 0.044194173824159216f;  // 1/sqrt(512)
    float a = modb[t];
    for (int k = 0; k < 512; k++) a += modw[t * 512 + k] * inv * s_code[k];
    style[b * 128 + t] = a;
  }
}

__global__ void k_modw(const float* __restrict__ scw, const float* __restrict__ style,
                       unsigned short* __restrict__ wmod) {
  __shared__ float red[128];
  const int co = blockIdx.x, b = blockIdx.y, ci = threadIdx.x;
  const float s1 = 0.029462782549439483f;  // 1/sqrt(128*9)
  const float st = style[b * 128 + ci];
  const float* base = scw + ((long)co * 128 + ci) * 9;
  float wloc[9]; float sq = 0.f;
#pragma unroll
  for (int j = 0; j < 9; j++) { const float v = base[j] * s1 * st; wloc[j] = v; sq += v * v; }
  red[ci] = sq; __syncthreads();
  for (int o = 64; o > 0; o >>= 1) {
    if (ci < o) red[ci] += red[ci + o];
    __syncthreads();
  }
  const float demod = rsqrtf(red[0] + 1e-8f);
  unsigned short* d = wmod + (((long)b * 128 + co) * 1152) + ci * 9;
#pragma unroll
  for (int j = 0; j < 9; j++) d[j] = f2bf(wloc[j] * demod);
}

__global__ void k_att(const float* __restrict__ avec, const float* __restrict__ w1,
                      const float* __restrict__ w2, const float* __restrict__ b2,
                      const float* __restrict__ dynb, float* __restrict__ attout,
                      float* __restrict__ aggb) {
  __shared__ float s_a[128], s_h[33], s_l[4], s_att[4];
  const int b = blockIdx.x, t = threadIdx.x;
  if (t < 128) s_a[t] = avec[b * 128 + t];
  __syncthreads();
  if (t < 33) {
    float a = 0.f;
    for (int k = 0; k < 128; k++) a += w1[t * 128 + k] * s_a[k];
    s_h[t] = fmaxf(a, 0.f);
  }
  __syncthreads();
  if (t < 4) {
    float a = b2[t];
    for (int k = 0; k < 33; k++) a += w2[t * 33 + k] * s_h[k];
    s_l[t] = a * (1.f / 34.f);
  }
  __syncthreads();
  if (t == 0) {
    float mx = s_l[0];
    for (int i = 1; i < 4; i++) mx = fmaxf(mx, s_l[i]);
    float sum = 0.f;
    for (int i = 0; i < 4; i++) { s_att[i] = __expf(s_l[i] - mx); sum += s_att[i]; }
    for (int i = 0; i < 4; i++) { s_att[i] /= sum; attout[b * 4 + i] = s_att[i]; }
  }
  __syncthreads();
  {
    float a = 0.f;
    for (int k = 0; k < 4; k++) a += s_att[k] * dynb[k * 256 + t];
    aggb[b * 256 + t] = a;
  }
}

__global__ void k_aggw(const float* __restrict__ att, const float* __restrict__ dynw,
                       unsigned short* __restrict__ aggw) {
  const int co = blockIdx.x, b = blockIdx.y, ci = threadIdx.x;
  const float a0 = att[b * 4 + 0], a1 = att[b * 4 + 1], a2 = att[b * 4 + 2], a3 = att[b * 4 + 3];
  const long o = ((long)co * 128 + ci) * 9;
  const long ks = 256L * 128 * 9;
  unsigned short* d = aggw + (((long)b * 256 + co) * 1152) + ci * 9;
#pragma unroll
  for (int j = 0; j < 9; j++) {
    const float v = a0 * dynw[o + j] + a1 * dynw[ks + o + j] + a2 * dynw[2 * ks + o + j]
                  + a3 * dynw[3 * ks + o + j];
    d[j] = f2bf(v);
  }
}

__global__ void k_pack_kv(const float* __restrict__ kv, unsigned short* __restrict__ ktok,
                          unsigned short* __restrict__ vmat) {
  const int b = blockIdx.y, n = blockIdx.x, c = threadIdx.x;
  ktok[((long)b * 4096 + n) * 128 + c] = f2bf(kv[(((long)b * 256 + c) * 4096) + n]);
  vmat[(((long)b * 128 + c) * 4096) + n] = f2bf(kv[(((long)b * 256 + 128 + c) * 4096) + n]);
}

__global__ void k_instnorm(const float* __restrict__ src, const float* __restrict__ g,
                           const float* __restrict__ be, float* __restrict__ out) {
  __shared__ float rs[256], rq[256];
  const int row = blockIdx.x, b = row >> 7, c = row & 127;
  const float* s = src + (long)row * 4096;
  float vloc[16]; float sum = 0.f, sq = 0.f;
#pragma unroll
  for (int i = 0; i < 16; i++) {
    const float v = s[threadIdx.x + i * 256];
    vloc[i] = v; sum += v; sq += v * v;
  }
  rs[threadIdx.x] = sum; rq[threadIdx.x] = sq; __syncthreads();
  for (int o = 128; o > 0; o >>= 1) {
    if ((int)threadIdx.x < o) { rs[threadIdx.x] += rs[threadIdx.x + o]; rq[threadIdx.x] += rq[threadIdx.x + o]; }
    __syncthreads();
  }
  const float mu = rs[0] * (1.f / 4096.f);
  const float var = rq[0] * (1.f / 4096.f) - mu * mu;
  const float inv = rsqrtf(var + 1e-5f);
  const float gg = g[c], bb = be[c];
  float* o_ = out + (((long)b * 256 + 2 * c + 1) * 4096);
#pragma unroll
  for (int i = 0; i < 16; i++)
    o_[threadIdx.x + i * 256] = fmaxf((vloc[i] - mu) * inv * gg + bb, 0.f);
}

// ---------------------------------------------------------------------------
// Host orchestration
// ---------------------------------------------------------------------------
extern "C" void kernel_launch(void* const* d_in, const int* in_sizes, int n_in,
                              void* d_out, int out_size, void* d_ws, size_t ws_size,
                              hipStream_t stream) {
  (void)in_sizes; (void)n_in; (void)out_size; (void)ws_size;
  const float* feature = (const float*)d_in[0];
  const float* keyp    = (const float*)d_in[1];
  const float* noise   = (const float*)d_in[2];
  const float* mb      = (const float*)d_in[3];
  const float* proj_w  = (const float*)d_in[4];
  const float* proj_b  = (const float*)d_in[5];
  const float* rw1     = (const float*)d_in[6];
  const float* rb1     = (const float*)d_in[7];
  const float* rw2     = (const float*)d_in[8];
  const float* rb2     = (const float*)d_in[9];
  const float* mod_w   = (const float*)d_in[10];
  const float* mod_b   = (const float*)d_in[11];
  const float* sc_w    = (const float*)d_in[12];
  const float* noise_w = (const float*)d_in[13];
  const float* act_b   = (const float*)d_in[14];
  const float* aw1     = (const float*)d_in[15];
  const float* aw2     = (const float*)d_in[16];
  const float* ab2     = (const float*)d_in[17];
  const float* dyn_w   = (const float*)d_in[18];
  const float* dyn_b   = (const float*)d_in[19];
  const float* q_w     = (const float*)d_in[20];
  const float* q_b     = (const float*)d_in[21];
  const float* diff_w  = (const float*)d_in[22];
  const float* diff_b  = (const float*)d_in[23];
  const float* c64_w   = (const float*)d_in[24];
  const float* c64_b   = (const float*)d_in[25];
  const float* in_g    = (const float*)d_in[26];
  const float* in_b    = (const float*)d_in[27];
  float* out = (float*)d_out;

  unsigned char* ws = (unsigned char*)d_ws;
  size_t off = 0;
  auto alloc = [&](size_t bytes) -> void* {
    void* p = ws + off;
    off += (bytes + 255) & ~(size_t)255;
    return p;
  };

  float*          feat     = (float*)          alloc(4UL * 128 * 4096 * 4);
  unsigned short* feat_tok = (unsigned short*) alloc(4UL * 4096 * 128 * 2);
  unsigned short* cstok    = (unsigned short*) alloc(4UL * 4096 * 128 * 2);
  unsigned short* projw_bf = (unsigned short*) alloc(128UL * 128 * 2);
  unsigned short* qw_bf    = (unsigned short*) alloc(128UL * 128 * 2);
  unsigned short* diffw_bf = (unsigned short*) alloc(128UL * 128 * 2);
  unsigned short* c64w_bf  = (unsigned short*) alloc(128UL * 1152 * 2);
  float*          scode    = (float*)          alloc(512 * 4);
  float*          scbuf    = (float*)          alloc(4UL * 158 * 4);
  float*          style    = (float*)          alloc(512 * 4);
  float*          avec     = (float*)          alloc(512 * 4);
  float*          attv     = (float*)          alloc(16 * 4);
  float*          aggb     = (float*)          alloc(1024 * 4);
  unsigned short* wmod     = (unsigned short*) alloc(4UL * 128 * 1152 * 2);
  unsigned short* bt_mb    = (unsigned short*) alloc(4096UL * 1152 * 2);
  float*          fbuf2    = (float*)          alloc(4UL * 128 * 4096 * 4);  // expand_mb / sum
  unsigned short* aggw     = (unsigned short*) alloc(4UL * 256 * 1152 * 2);
  unsigned short* bt_ic    = (unsigned short*) alloc(4UL * 4096 * 1152 * 2);
  float*          kvbuf    = (float*)          alloc(4UL * 256 * 4096 * 4);  // kv / conv64 out
  unsigned short* qtok     = (unsigned short*) alloc(4UL * 4096 * 128 * 2);
  unsigned short* ktok     = (unsigned short*) alloc(4UL * 4096 * 128 * 2);
  unsigned short* vmat     = (unsigned short*) alloc(4UL * 4096 * 128 * 2);
  unsigned short* diffin   = (unsigned short*) alloc(4UL * 4096 * 128 * 2);

  // 1) even channels pass through; pack odd channels as tokens; pack static weights
  k_copy_even<<<8192, 256, 0, stream>>>(feature, out);
  k_f2bf<<<64, 256, 0, stream>>>(proj_w, projw_bf, 16384);
  k_f2bf<<<64, 256, 0, stream>>>(q_w, qw_bf, 16384);
  k_f2bf<<<64, 256, 0, stream>>>(diff_w, diffw_bf, 16384);
  k_f2bf<<<576, 256, 0, stream>>>(c64_w, c64w_bf, 147456);
  k_pack_cs<<<dim3(4096, 4), 128, 0, stream>>>(feature, cstok);

  // 2) feat = proj(out_cs) + b   (also emit feat tokens for the q-conv Bt)
  k_gemm<<<dim3(64, 2, 4), 128, 0, stream>>>(projw_bf, 0, cstok, 4096L * 128, proj_b, 0, 0,
                                             nullptr, nullptr, nullptr, feat, feat_tok,
                                             128, 4096, 128);

  // 3) routing MLP -> style; modulated+demodulated per-sample weights
  k_rowmean<<<512, 256, 0, stream>>>(feat, scode);
  k_build_sc<<<4, 160, 0, stream>>>(scode, keyp, scbuf);
  k_mlp<<<4, 512, 0, stream>>>(scbuf, rw1, rb1, rw2, rb2, mod_w, mod_b, style);
  k_modw<<<dim3(128, 4), 128, 0, stream>>>(sc_w, style, wmod);

  // 4) expand_mb = per-sample 3x3 conv on shared mb; noise + act_b + leaky*sqrt2
  k_im2col<<<dim3(4096, 1, 1), 128, 0, stream>>>(mb, 0, bt_mb, 0, 128);
  k_gemm<<<dim3(64, 2, 4), 128, 0, stream>>>(wmod, 128L * 1152, bt_mb, 0, act_b, 0, 2,
                                             noise, noise_w, nullptr, fbuf2, nullptr,
                                             128, 4096, 1152);
  k_rowmean<<<512, 256, 0, stream>>>(fbuf2, avec);

  // 5) attention routing -> dynamic weights/bias; kv = relu(dynconv(feat)); q = relu(1x1(feat))
  k_att<<<4, 256, 0, stream>>>(avec, aw1, aw2, ab2, dyn_b, attv, aggb);
  k_aggw<<<dim3(256, 4), 128, 0, stream>>>(attv, dyn_w, aggw);
  k_im2col<<<dim3(4096, 1, 4), 128, 0, stream>>>(feat, 128L * 4096, bt_ic, 4096L * 1152, 128);
  k_gemm<<<dim3(64, 4, 4), 128, 0, stream>>>(aggw, 256L * 1152, bt_ic, 4096L * 1152, aggb, 256, 1,
                                             nullptr, nullptr, nullptr, kvbuf, nullptr,
                                             256, 4096, 1152);
  k_gemm<<<dim3(64, 2, 4), 128, 0, stream>>>(qw_bf, 0, feat_tok, 4096L * 128, q_b, 0, 1,
                                             nullptr, nullptr, nullptr, nullptr, qtok,
                                             128, 4096, 128);
  k_pack_kv<<<dim3(4096, 4), 128, 0, stream>>>(kvbuf, ktok, vmat);

  // 6) flash attention -> diffin = feat - residual (tokens)
  k_attn<<<dim3(64, 4), 128, 0, stream>>>(qtok, ktok, vmat, feat, diffin);

  // 7) sum = relu(diff_conv(diffin)) + feat ; conv64 3x3 ; instance norm + relu -> odd channels
  k_gemm<<<dim3(64, 2, 4), 128, 0, stream>>>(diffw_bf, 0, diffin, 4096L * 128, diff_b, 0, 1,
                                             nullptr, nullptr, feat, fbuf2, nullptr,
                                             128, 4096, 128);
  k_im2col<<<dim3(4096, 1, 4), 128, 0, stream>>>(fbuf2, 128L * 4096, bt_ic, 4096L * 1152, 128);
  k_gemm<<<dim3(64, 2, 4), 128, 0, stream>>>(c64w_bf, 0, bt_ic, 4096L * 1152, c64_b, 0, 0,
                                             nullptr, nullptr, nullptr, kvbuf, nullptr,
                                             128, 4096, 1152);
  k_instnorm<<<512, 256, 0, stream>>>(kvbuf, in_g, in_b, out);
}
